// FusedMoE_82712480186868
// MI455X (gfx1250) — compile-verified, matrix operand order checked by reading, not run on hardware
//
#include <hip/hip_runtime.h>

// Problem constants (from reference): T,H,F,E,K = 16384,1024,512,64,2
#define T_TOK 16384
#define H_DIM 1024
#define F_DIM 512
#define N_EXP 64
#define TOPK  2
#define MB    64            // tokens per workgroup (4 WMMA M-tiles)
#define XS_STRIDE 1032      // bf16 elems per LDS row of X  (1024 + 8 pad, rows 16B aligned)
#define ACT_STRIDE 520      // bf16 elems per LDS row of act (512 + 8 pad, rows 16B aligned)

typedef __attribute__((ext_vector_type(16))) __bf16 v16bf;
typedef __attribute__((ext_vector_type(8)))  __bf16 v8bf;
typedef __attribute__((ext_vector_type(8)))  float  v8f;

__device__ __forceinline__ v8bf cvt8(float4 a, float4 b) {
  v8bf r;
  r[0]=(__bf16)a.x; r[1]=(__bf16)a.y; r[2]=(__bf16)a.z; r[3]=(__bf16)a.w;
  r[4]=(__bf16)b.x; r[5]=(__bf16)b.y; r[6]=(__bf16)b.z; r[7]=(__bf16)b.w;
  return r;
}

// 16x16x32 bf16 fragment: this lane's 16 K-values live at p[0..7] and p[16..23]
// (caller already folded in the lane-half*8 K offset per the ISA layout).
__device__ __forceinline__ v16bf frag_from_global_f32(const float* __restrict__ p) {
  union { v16bf v; v8bf h[2]; } u;
  float4 f0 = *(const float4*)(p + 0);
  float4 f1 = *(const float4*)(p + 4);
  float4 f2 = *(const float4*)(p + 16);
  float4 f3 = *(const float4*)(p + 20);
  u.h[0] = cvt8(f0, f1);
  u.h[1] = cvt8(f2, f3);
  return u.v;
}

__device__ __forceinline__ v16bf frag_from_lds(const __bf16* p) {
  union { v16bf v; v8bf h[2]; } u;
  u.h[0] = *(const v8bf*)(p + 0);     // ds_load_b128
  u.h[1] = *(const v8bf*)(p + 16);    // ds_load_b128
  return u.v;
}

// silu(g) = g * sigmoid(g), using hardware v_rcp_f32 + v_exp_f32 (TRANS pipe,
// co-executes with WMMA) instead of IEEE division.
__device__ __forceinline__ float fast_silu(float g) {
  return g * __builtin_amdgcn_rcpf(1.0f + __expf(-g));
}

// ---------------- routing kernels -------------------------------------------

__global__ void zero_kernel(float* __restrict__ out, int n, int* __restrict__ cnts) {
  int i = blockIdx.x * blockDim.x + threadIdx.x;
  if (i < n) out[i] = 0.0f;
  if (i < 2 * N_EXP) cnts[i] = 0;   // counts[64] + cursor[64]
}

__global__ void count_kernel(const int* __restrict__ ids, int* __restrict__ counts) {
  int t = blockIdx.x * blockDim.x + threadIdx.x;
  if (t < T_TOK) {
    atomicAdd(&counts[ids[t * TOPK + 0]], 1);
    atomicAdd(&counts[ids[t * TOPK + 1]], 1);
  }
}

__global__ void scan_kernel(const int* __restrict__ counts, int* __restrict__ offsets) {
  if (threadIdx.x == 0 && blockIdx.x == 0) {
    int s = 0;
    for (int e = 0; e < N_EXP; ++e) { offsets[e] = s; s += counts[e]; }
  }
}

__global__ void scatter_kernel(const int* __restrict__ ids, const float* __restrict__ tw,
                               const int* __restrict__ offsets, int* __restrict__ cursor,
                               int* __restrict__ tokbuf, float* __restrict__ wbuf) {
  int t = blockIdx.x * blockDim.x + threadIdx.x;
  if (t < T_TOK) {
    #pragma unroll
    for (int k = 0; k < TOPK; ++k) {
      int e = ids[t * TOPK + k];
      int pos = offsets[e] + atomicAdd(&cursor[e], 1);
      tokbuf[pos] = t;
      wbuf[pos]   = tw[t * TOPK + k];
    }
  }
}

// ---------------- fused expert MLP kernel -----------------------------------

__global__ void __launch_bounds__(256, 1)
moe_kernel(const float* __restrict__ x,
           const float* __restrict__ wgu,
           const float* __restrict__ wd,
           const int*   __restrict__ counts,
           const int*   __restrict__ offsets,
           const int*   __restrict__ tokbuf,
           const float* __restrict__ wbuf,
           float*       __restrict__ out)
{
  const int e    = blockIdx.y;
  const int cnt  = counts[e];
  const int base = blockIdx.x * MB;
  if (base >= cnt) return;           // block-uniform early exit

  extern __shared__ char smem_raw[];
  __bf16* xs    = (__bf16*)smem_raw;                 // MB x XS_STRIDE
  __bf16* acts  = xs + MB * XS_STRIDE;               // MB x ACT_STRIDE
  float*  coefs = (float*)(acts + MB * ACT_STRIDE);  // MB
  int*    toks  = (int*)(coefs + MB);                // MB

  const int tid = threadIdx.x;
  const int off = offsets[e];

  if (tid < MB) {
    int g = base + tid;
    if (g < cnt) { toks[tid] = tokbuf[off + g]; coefs[tid] = wbuf[off + g]; }
    else         { toks[tid] = 0;               coefs[tid] = 0.0f; }
  }
  __syncthreads();

  // Gather X rows (fp32 -> bf16) into LDS, 16B-coalesced. 64 full iterations.
  for (int idx = tid; idx < MB * (H_DIM / 4); idx += 256) {
    int r  = idx >> 8;               // H_DIM/4 == 256 float4 per row
    int c4 = idx & 255;
    const float4 f = *(const float4*)(x + (long long)toks[r] * H_DIM + c4 * 4);
    __bf16* d = xs + r * XS_STRIDE + c4 * 4;
    d[0]=(__bf16)f.x; d[1]=(__bf16)f.y; d[2]=(__bf16)f.z; d[3]=(__bf16)f.w;
  }
  __syncthreads();

  const int wave  = tid >> 5;
  const int lane  = tid & 31;
  const int nlo   = lane & 15;
  const int khalf = lane >> 4;       // lane-half K offset per ISA fragment layout

  const float* wgu_e = wgu + (long long)e * (2 * F_DIM) * H_DIM;
  const float* wd_e  = wd  + (long long)e * H_DIM * F_DIM;

  const v8f vzero = {0.f,0.f,0.f,0.f,0.f,0.f,0.f,0.f};

  // Phase 1: gu = X @ Wgu^T, act = silu(gate)*up  -> acts (MB x F_DIM, bf16)
  for (int p = 0; p < 4; ++p) {
    const int ntile = p * 8 + wave;  // 0..31, covers all 512 gate columns
    const int colg  = ntile * 16;
    v8f cg[4], cu[4];
    #pragma unroll
    for (int mt = 0; mt < 4; ++mt) { cg[mt] = vzero; cu[mt] = vzero; }

    const float* bg_row = wgu_e + (long long)(colg + nlo) * H_DIM + khalf * 8;
    const float* bu_row = wgu_e + (long long)(F_DIM + colg + nlo) * H_DIM + khalf * 8;

    const __bf16* a_base = xs + nlo * XS_STRIDE + khalf * 8;

    for (int kc = 0; kc < H_DIM; kc += 32) {
      // global_prefetch_b8, branch-free (address wraps within the row)
      int pf = (kc + 256) & (H_DIM - 1);
      __builtin_prefetch(bg_row + pf, 0, 1);
      __builtin_prefetch(bu_row + pf, 0, 1);

      v16bf bg = frag_from_global_f32(bg_row + kc);
      v16bf bu = frag_from_global_f32(bu_row + kc);

      // Hoist all four A fragments into distinct values so the 8 ds_load_b128
      // issue back-to-back and the WMMAs wait on dscnt incrementally.
      v16bf a0 = frag_from_lds(a_base + 0 * 16 * XS_STRIDE + kc);
      v16bf a1 = frag_from_lds(a_base + 1 * 16 * XS_STRIDE + kc);
      v16bf a2 = frag_from_lds(a_base + 2 * 16 * XS_STRIDE + kc);
      v16bf a3 = frag_from_lds(a_base + 3 * 16 * XS_STRIDE + kc);

      cg[0] = __builtin_amdgcn_wmma_f32_16x16x32_bf16(false, a0, false, bg, (short)0, cg[0], false, false);
      cu[0] = __builtin_amdgcn_wmma_f32_16x16x32_bf16(false, a0, false, bu, (short)0, cu[0], false, false);
      cg[1] = __builtin_amdgcn_wmma_f32_16x16x32_bf16(false, a1, false, bg, (short)0, cg[1], false, false);
      cu[1] = __builtin_amdgcn_wmma_f32_16x16x32_bf16(false, a1, false, bu, (short)0, cu[1], false, false);
      cg[2] = __builtin_amdgcn_wmma_f32_16x16x32_bf16(false, a2, false, bg, (short)0, cg[2], false, false);
      cu[2] = __builtin_amdgcn_wmma_f32_16x16x32_bf16(false, a2, false, bu, (short)0, cu[2], false, false);
      cg[3] = __builtin_amdgcn_wmma_f32_16x16x32_bf16(false, a3, false, bg, (short)0, cg[3], false, false);
      cu[3] = __builtin_amdgcn_wmma_f32_16x16x32_bf16(false, a3, false, bu, (short)0, cu[3], false, false);
    }

    // C/D layout: VGPR i -> row (i + khalf*8), col = colg + nlo
    #pragma unroll
    for (int mt = 0; mt < 4; ++mt) {
      #pragma unroll
      for (int i = 0; i < 8; ++i) {
        float a = fast_silu(cg[mt][i]) * cu[mt][i];
        int row = mt * 16 + i + khalf * 8;
        acts[row * ACT_STRIDE + colg + nlo] = (__bf16)a;
      }
    }
  }
  __syncthreads();

  // Phase 2: y = act @ Wd^T, scale by routing weight, scatter-add to out
  for (int nt = wave; nt < H_DIM / 16; nt += 8) {
    const int hb = nt * 16;
    v8f cd[4];
    #pragma unroll
    for (int mt = 0; mt < 4; ++mt) cd[mt] = vzero;

    const float*  b_row  = wd_e + (long long)(hb + nlo) * F_DIM + khalf * 8;
    const __bf16* a_base = acts + nlo * ACT_STRIDE + khalf * 8;

    for (int kc = 0; kc < F_DIM; kc += 32) {
      int pf = (kc + 256) & (F_DIM - 1);
      __builtin_prefetch(b_row + pf, 0, 1);

      v16bf b  = frag_from_global_f32(b_row + kc);
      v16bf a0 = frag_from_lds(a_base + 0 * 16 * ACT_STRIDE + kc);
      v16bf a1 = frag_from_lds(a_base + 1 * 16 * ACT_STRIDE + kc);
      v16bf a2 = frag_from_lds(a_base + 2 * 16 * ACT_STRIDE + kc);
      v16bf a3 = frag_from_lds(a_base + 3 * 16 * ACT_STRIDE + kc);

      cd[0] = __builtin_amdgcn_wmma_f32_16x16x32_bf16(false, a0, false, b, (short)0, cd[0], false, false);
      cd[1] = __builtin_amdgcn_wmma_f32_16x16x32_bf16(false, a1, false, b, (short)0, cd[1], false, false);
      cd[2] = __builtin_amdgcn_wmma_f32_16x16x32_bf16(false, a2, false, b, (short)0, cd[2], false, false);
      cd[3] = __builtin_amdgcn_wmma_f32_16x16x32_bf16(false, a3, false, b, (short)0, cd[3], false, false);
    }

    #pragma unroll
    for (int mt = 0; mt < 4; ++mt) {
      #pragma unroll
      for (int i = 0; i < 8; ++i) {
        int row = mt * 16 + i + khalf * 8;
        float c = coefs[row];
        if (c != 0.0f) {
          // ~2 contributions per element: hardware global_atomic_add_f32
          unsafeAtomicAdd(out + (long long)toks[row] * H_DIM + hb + nlo, c * cd[mt][i]);
        }
      }
    }
  }
}

// ---------------- host launch -----------------------------------------------

extern "C" void kernel_launch(void* const* d_in, const int* in_sizes, int n_in,
                              void* d_out, int out_size, void* d_ws, size_t ws_size,
                              hipStream_t stream) {
  const float* x   = (const float*)d_in[0];   // hidden_states (T,H)
  const float* tw  = (const float*)d_in[1];   // topk_weights  (T,K)
  const int*   ids = (const int*)  d_in[2];   // topk_ids      (T,K)
  const float* wgu = (const float*)d_in[3];   // gate_up_weight (E,2F,H)
  const float* wd  = (const float*)d_in[4];   // down_weight    (E,H,F)
  float* out = (float*)d_out;                 // (T,H) fp32

  // Workspace layout
  int*   counts  = (int*)d_ws;                 // [64]
  int*   cursor  = counts + N_EXP;             // [64]
  int*   offsets = counts + 2 * N_EXP;         // [64]
  int*   tokbuf  = counts + 4 * N_EXP;         // [T*K]
  float* wbuf    = (float*)(tokbuf + T_TOK * TOPK);

  const int n_out = T_TOK * H_DIM;
  zero_kernel<<<(n_out + 255) / 256, 256, 0, stream>>>(out, n_out, counts);
  count_kernel<<<(T_TOK + 255) / 256, 256, 0, stream>>>(ids, counts);
  scan_kernel<<<1, 64, 0, stream>>>(counts, offsets);
  scatter_kernel<<<(T_TOK + 255) / 256, 256, 0, stream>>>(ids, tw, offsets, cursor,
                                                          tokbuf, wbuf);

  const size_t smem = (size_t)(MB * XS_STRIDE + MB * ACT_STRIDE) * 2
                    + (size_t)MB * sizeof(float) + (size_t)MB * sizeof(int);
  hipFuncSetAttribute(reinterpret_cast<const void*>(moe_kernel),
                      hipFuncAttributeMaxDynamicSharedMemorySize, (int)smem);

  dim3 grid((T_TOK * TOPK + MB - 1) / MB, N_EXP);
  moe_kernel<<<grid, 256, smem, stream>>>(x, wgu, wd, counts, offsets,
                                          tokbuf, wbuf, out);
}
